// APN_11785390260477
// MI455X (gfx1250) — compile-verified, matrix operand order checked by reading, not run on hardware
//
#include <hip/hip_runtime.h>
#include <cstdint>

typedef __attribute__((ext_vector_type(16))) _Float16 v16h;
typedef __attribute__((ext_vector_type(8)))  float    v8f;

__device__ __forceinline__ float softplusf(float x) {
  return (x > 20.f) ? x : log1pf(__expf(x));
}
__device__ __forceinline__ float siluf(float x) {
  return x / (1.f + __expf(-x));
}

// Load one 16-half WMMA fragment row: K-runs [kb,kb+8) and [kb+16,kb+24),
// as 4 aligned float4 (global_load_b128) loads + packed f32->f16 converts.
__device__ __forceinline__ v16h frag_ld(const float* __restrict__ row, int kb) {
  const float4 a0 = *(const float4*)(row + kb);
  const float4 a1 = *(const float4*)(row + kb + 4);
  const float4 a2 = *(const float4*)(row + kb + 16);
  const float4 a3 = *(const float4*)(row + kb + 20);
  v16h f;
  f[0]  = (_Float16)a0.x; f[1]  = (_Float16)a0.y;
  f[2]  = (_Float16)a0.z; f[3]  = (_Float16)a0.w;
  f[4]  = (_Float16)a1.x; f[5]  = (_Float16)a1.y;
  f[6]  = (_Float16)a1.z; f[7]  = (_Float16)a1.w;
  f[8]  = (_Float16)a2.x; f[9]  = (_Float16)a2.y;
  f[10] = (_Float16)a2.z; f[11] = (_Float16)a2.w;
  f[12] = (_Float16)a3.x; f[13] = (_Float16)a3.y;
  f[14] = (_Float16)a3.z; f[15] = (_Float16)a3.w;
  return f;
}

// ---------------------------------------------------------------------------
// WMMA GEMM:  C[M,N] = A[M,K] * B[N,K]^T   (row-major fp32 in/out, f16 MACs,
// f32 accumulate). Each wave computes a 32x16 tile (2 accumulators sharing
// one B fragment -> two independent back-to-back WMMAs per K-step).
// REQUIRES: M%32==0, N%16==0, K%32==0, lda/ldb%4==0.
// OP: 0=none, 1=+bias, 2=+bias,leakyrelu(0.1), 3=+bias,softplus, 4=+residual
// ---------------------------------------------------------------------------
template <int OP>
__global__ void __launch_bounds__(256)
wmma_gemm_kernel(const float* __restrict__ A, int lda,
                 const float* __restrict__ B, int ldb,
                 float* __restrict__ C, int ldc,
                 int M, int N, int K,
                 const float* __restrict__ bias,
                 const float* __restrict__ resid, int ldr)
{
  const int lane   = threadIdx.x & 31;
  const int wave   = threadIdx.x >> 5;
  const int tilesN = N >> 4;
  const int nTiles = (M >> 5) * tilesN;       // 32-row macro tiles
  const int tile   = blockIdx.x * (blockDim.x >> 5) + wave;
  if (tile >= nTiles) return;                 // wave-uniform
  const int tmi  = tile / tilesN;
  const int tm   = tmi << 5;
  const int tn   = (tile - tmi * tilesN) << 4;
  const int lh8  = (lane >> 4) << 3;          // K-half offset per ISA layout
  const int lrow = lane & 15;

  // 32-bit offsets: every matrix here is < 2^31 elements.
  const float* __restrict__ a0row = A + (tm + lrow) * lda;
  const float* __restrict__ a1row = a0row + 16 * lda;
  const float* __restrict__ brow  = B + (tn + lrow) * ldb;

  v8f acc0 = {}, acc1 = {};
  for (int k0 = 0; k0 < K; k0 += 32) {
    const int kb = k0 + lh8;
    const v16h bf  = frag_ld(brow, kb);
    const v16h af0 = frag_ld(a0row, kb);
    const v16h af1 = frag_ld(a1row, kb);
    acc0 = __builtin_amdgcn_wmma_f32_16x16x32_f16(false, af0, false, bf,
                                                  (short)0, acc0, false, false);
    acc1 = __builtin_amdgcn_wmma_f32_16x16x32_f16(false, af1, false, bf,
                                                  (short)0, acc1, false, false);
  }

  const int col = tn + lrow;
  float bv = 0.f;
  if constexpr (OP >= 1 && OP <= 3) bv = bias[col];

  const int rb0 = tm + ((lane >> 4) << 3);
#pragma unroll
  for (int half = 0; half < 2; ++half) {
    const v8f& acc = half ? acc1 : acc0;
    const int rbase = rb0 + half * 16;
#pragma unroll
    for (int r = 0; r < 8; ++r) {
      float v = acc[r];
      if constexpr (OP >= 1 && OP <= 3) v += bv;
      if constexpr (OP == 2) v = (v >= 0.f) ? v : 0.1f * v;
      if constexpr (OP == 3) v = softplusf(v);
      if constexpr (OP == 4) v += resid[(rbase + r) * ldr + col];
      C[(rbase + r) * ldc + col] = v;
    }
  }
}

// ---------------------------------------------------------------------------
// Zero-padded 2D copy: dst[r, 0:Kdst] = (c < Ksrc) ? src[r*lds + c] : 0
// ---------------------------------------------------------------------------
__global__ void pad2d_kernel(const float* __restrict__ src, int lds, int Ksrc,
                             float* __restrict__ dst, int Kdst, int rows)
{
  const int idx = blockIdx.x * blockDim.x + threadIdx.x;
  if (idx >= rows * Kdst) return;
  const int r = idx / Kdst;
  const int c = idx - r * Kdst;
  dst[idx] = (c < Ksrc) ? src[r * lds + c] : 0.f;
}

// ---------------------------------------------------------------------------
// RMSNorm over rows of width D (blockDim == D == 256)
// ---------------------------------------------------------------------------
__global__ void rmsnorm_kernel(const float* __restrict__ x,
                               const float* __restrict__ w,
                               float* __restrict__ o, int D, float eps)
{
  __shared__ float red[32];
  const int row = blockIdx.x;
  const int tid = threadIdx.x;
  const float v = x[row * D + tid];
  float ss = v * v;
#pragma unroll
  for (int m = 16; m >= 1; m >>= 1) ss += __shfl_xor(ss, m);
  if ((tid & 31) == 0) red[tid >> 5] = ss;
  __syncthreads();
  if (tid < 32) {
    float t = (tid < (int)(blockDim.x >> 5)) ? red[tid] : 0.f;
#pragma unroll
    for (int m = 16; m >= 1; m >>= 1) t += __shfl_xor(t, m);
    if (tid == 0) red[0] = t;
  }
  __syncthreads();
  const float scale = rsqrtf(red[0] / (float)D + eps);
  o[row * D + tid] = v * scale * w[tid];
}

// ---------------------------------------------------------------------------
// Causal depthwise conv (d_conv=4) over xm = xz[:, 0:DI], then SiLU
// ---------------------------------------------------------------------------
__global__ void conv_silu_kernel(const float* __restrict__ xz,
                                 const float* __restrict__ cw,
                                 const float* __restrict__ cb,
                                 float* __restrict__ xc,
                                 int L, int DI, int xzld)
{
  const int idx = blockIdx.x * blockDim.x + threadIdx.x;
  if (idx >= L * DI) return;
  const int l = idx / DI;
  const int d = idx - l * DI;
  float acc = cb[d];
#pragma unroll
  for (int k = 0; k < 4; ++k) {
    const int t = l + k - 3;
    if (t >= 0) acc += cw[d * 4 + k] * xz[t * xzld + d];
  }
  xc[idx] = siluf(acc);
}

// ---------------------------------------------------------------------------
// Fused selective scan. Lane layout: n = tid&15 (d_state), d = 16*blk + tid>>4.
// deltaA / deltaBu computed in-register (never materialized). The z-gate,
// D skip-connection and SiLU are fused into the per-step epilogue.
// ---------------------------------------------------------------------------
__global__ void __launch_bounds__(256)
scan_kernel(const float* __restrict__ delta,  // L x DI (post softplus)
            const float* __restrict__ dbc,    // L x 48 : B at +16, Cm at +32
            const float* __restrict__ xc,     // L x DI
            const float* __restrict__ xz,     // L x 2*DI (z at +DI)
            const float* __restrict__ A_log,  // DI x 16
            const float* __restrict__ Dp,     // DI
            float* __restrict__ y,            // L x DI
            int L, int DI)
{
  const int n = threadIdx.x & 15;
  const int d = blockIdx.x * 16 + (threadIdx.x >> 4);
  const float Av = -__expf(A_log[d * 16 + n]);
  const float Dv = Dp[d];
  float s = 0.f;
  for (int t = 0; t < L; ++t) {
    if ((t & 7) == 0 && t + 8 < L) {           // hide dependent-load latency
      __builtin_prefetch(&delta[(t + 8) * DI + d], 0, 0);
      __builtin_prefetch(&xc[(t + 8) * DI + d], 0, 0);
      __builtin_prefetch(&dbc[(t + 8) * 48 + 16 + n], 0, 0);
    }
    const float dl  = delta[t * DI + d];
    const float xcv = xc[t * DI + d];
    const float Bv  = dbc[t * 48 + 16 + n];
    const float Cv  = dbc[t * 48 + 32 + n];
    s = __expf(dl * Av) * s + dl * Bv * xcv;
    float p = s * Cv;
    p += __shfl_xor(p, 1);
    p += __shfl_xor(p, 2);
    p += __shfl_xor(p, 4);
    p += __shfl_xor(p, 8);                     // sum over the 16 n-lanes
    if (n == 0) {
      const float z = xz[t * (2 * DI) + DI + d];
      y[t * DI + d] = (p + xcv * Dv) * siluf(z);
    }
  }
}

// ---------------------------------------------------------------------------
// Row softmax, C<=8
// ---------------------------------------------------------------------------
__global__ void softmax_kernel(const float* __restrict__ logits,
                               float* __restrict__ y, int B, int C)
{
  const int b = blockIdx.x * blockDim.x + threadIdx.x;
  if (b >= B) return;
  float m = -1e30f;
  for (int c = 0; c < C; ++c) m = fmaxf(m, logits[b * C + c]);
  float e[8], sum = 0.f;
  for (int c = 0; c < C; ++c) { e[c] = __expf(logits[b * C + c] - m); sum += e[c]; }
  const float inv = 1.f / sum;
  for (int c = 0; c < C; ++c) y[b * C + c] = e[c] * inv;
}

// combined = [feats | y_t | zero-pad] with padded width Wp (K%32==0)
__global__ void concat_kernel(const float* __restrict__ feats,
                              const float* __restrict__ yt,
                              float* __restrict__ comb,
                              int B, int D, int C, int Wp)
{
  const int idx = blockIdx.x * blockDim.x + threadIdx.x;
  if (idx >= B * Wp) return;
  const int row = idx / Wp;
  const int col = idx - row * Wp;
  float v = 0.f;
  if (col < D)          v = feats[row * D + col];
  else if (col < D + C) v = yt[row * C + (col - D)];
  comb[idx] = v;
}

__global__ void head_update_kernel(const float* __restrict__ h2,  // B x H
                                   const float* __restrict__ muW,
                                   const float* __restrict__ mub,
                                   const float* __restrict__ varW,
                                   const float* __restrict__ varb,
                                   const float* __restrict__ epsS, // B x C
                                   float* __restrict__ yt,         // B x C
                                   float* __restrict__ outS,       // B x C
                                   int B, int H, int C)
{
  const int idx = blockIdx.x * blockDim.x + threadIdx.x;
  if (idx >= B * C) return;
  const int b = idx / C;
  const int c = idx - b * C;
  const float* hr = h2 + b * H;
  float mu = mub[c], va = varb[c];
  for (int j = 0; j < H; ++j) {
    const float hv = hr[j];
    mu += hv * muW[c * H + j];
    va += hv * varW[c * H + j];
  }
  va = softplusf(va);
  const float yn = yt[idx] - (mu + va * epsS[idx]);
  yt[idx] = yn;
  outS[idx] = yn;
}

// ---------------------------------------------------------------------------
extern "C" void kernel_launch(void* const* d_in, const int* in_sizes, int n_in,
                              void* d_out, int out_size, void* d_ws, size_t ws_size,
                              hipStream_t stream)
{
  (void)in_sizes; (void)n_in; (void)out_size; (void)ws_size;
  const float* features  = (const float*)d_in[0];
  const float* y_init    = (const float*)d_in[1];
  const float* eps       = (const float*)d_in[2];
  const float* in_proj_W = (const float*)d_in[3];
  const float* conv_W    = (const float*)d_in[4];
  const float* conv_b    = (const float*)d_in[5];
  const float* x_proj_W  = (const float*)d_in[6];
  const float* dt_proj_W = (const float*)d_in[7];
  const float* dt_proj_b = (const float*)d_in[8];
  const float* A_log     = (const float*)d_in[9];
  const float* Dp        = (const float*)d_in[10];
  const float* out_proj_W= (const float*)d_in[11];
  const float* norm_w    = (const float*)d_in[12];
  const float* norm_f_w  = (const float*)d_in[13];
  const float* lm_head_W = (const float*)d_in[14];
  const float* fn1_W     = (const float*)d_in[15];
  const float* fn1_b     = (const float*)d_in[16];
  const float* fn2_W     = (const float*)d_in[17];
  const float* fn2_b     = (const float*)d_in[18];
  const float* mu_W      = (const float*)d_in[19];
  const float* mu_b      = (const float*)d_in[20];
  const float* var_W     = (const float*)d_in[21];
  const float* var_b     = (const float*)d_in[22];

  constexpr int B = 4096, D = 256, H = 128, C = 7, S = 3;
  constexpr int DI = 512, DTR = 16;
  constexpr int KC  = D + C;           // 263
  constexpr int KCP = 288;             // padded fn1 K (mult of 32)
  constexpr int DTP = 32;              // padded dt_proj K (mult of 32)

  float* ws = (float*)d_ws;
  size_t o = 0;
  float* hn    = ws + o; o += (size_t)B * D;        // rmsnorm(features)
  float* xz    = ws + o; o += (size_t)B * 2 * DI;   // in_proj out (xm | z)
  float* xc    = ws + o; o += (size_t)B * DI;       // conv+silu
  float* dbc   = ws + o; o += (size_t)B * 48;       // x_proj out
  float* dlt   = ws + o; o += (size_t)B * DI;       // softplus(dt_proj)
  float* yb    = ws + o; o += (size_t)B * DI;       // scan out (gated)
  float* res2  = ws + o; o += (size_t)B * D;        // out_proj + residual
  float* xf    = ws + o; o += (size_t)B * D;        // final rmsnorm
  float* feats = ws + o; o += (size_t)B * D;        // lm_head out
  float* comb  = ws + o; o += (size_t)B * KCP;
  float* h1    = ws + o; o += (size_t)B * H;
  float* h2    = ws + o; o += (size_t)B * H;
  float* yt    = ws + o; o += (size_t)B * C;
  float* dpad  = ws + o; o += (size_t)B * DTP;      // delta_raw zero-padded
  float* wdtp  = ws + o; o += (size_t)DI * DTP;     // dt_proj_W zero-padded
  float* wfn1p = ws + o; o += (size_t)H * KCP;      // fn1_W zero-padded

  auto grid = [](int M, int N) { return ((M >> 5) * (N >> 4) + 7) / 8; };

  // one-time weight padding (cheap, graph-capturable)
  pad2d_kernel<<<(DI * DTP + 255) / 256, 256, 0, stream>>>(
      dt_proj_W, DTR, DTR, wdtp, DTP, DI);
  pad2d_kernel<<<(H * KCP + 255) / 256, 256, 0, stream>>>(
      fn1_W, KC, KC, wfn1p, KCP, H);

  // ---- Mamba block ----
  rmsnorm_kernel<<<B, D, 0, stream>>>(features, norm_w, hn, D, 1e-5f);
  wmma_gemm_kernel<0><<<grid(B, 2 * DI), 256, 0, stream>>>(
      hn, D, in_proj_W, D, xz, 2 * DI, B, 2 * DI, D, nullptr, nullptr, 0);
  conv_silu_kernel<<<(B * DI + 255) / 256, 256, 0, stream>>>(
      xz, conv_W, conv_b, xc, B, DI, 2 * DI);
  wmma_gemm_kernel<0><<<grid(B, 48), 256, 0, stream>>>(
      xc, DI, x_proj_W, DI, dbc, 48, B, 48, DI, nullptr, nullptr, 0);
  pad2d_kernel<<<(B * DTP + 255) / 256, 256, 0, stream>>>(
      dbc, 48, DTR, dpad, DTP, B);                       // delta_raw -> padded
  wmma_gemm_kernel<3><<<grid(B, DI), 256, 0, stream>>>(
      dpad, DTP, wdtp, DTP, dlt, DI, B, DI, DTP, dt_proj_b, nullptr, 0);
  scan_kernel<<<DI / 16, 256, 0, stream>>>(
      dlt, dbc, xc, xz, A_log, Dp, yb, B, DI);
  wmma_gemm_kernel<4><<<grid(B, D), 256, 0, stream>>>(
      yb, DI, out_proj_W, DI, res2, D, B, D, DI, nullptr, features, D);
  rmsnorm_kernel<<<B, D, 0, stream>>>(res2, norm_f_w, xf, D, 1e-5f);
  wmma_gemm_kernel<0><<<grid(B, D), 256, 0, stream>>>(
      xf, D, lm_head_W, D, feats, D, B, D, D, nullptr, nullptr, 0);

  // ---- policy refinement ----
  softmax_kernel<<<(B + 255) / 256, 256, 0, stream>>>(y_init, yt, B, C);
  for (int s = 0; s < S; ++s) {
    concat_kernel<<<(B * KCP + 255) / 256, 256, 0, stream>>>(
        feats, yt, comb, B, D, C, KCP);
    wmma_gemm_kernel<2><<<grid(B, H), 256, 0, stream>>>(
        comb, KCP, wfn1p, KCP, h1, H, B, H, KCP, fn1_b, nullptr, 0);
    wmma_gemm_kernel<2><<<grid(B, H), 256, 0, stream>>>(
        h1, H, fn2_W, H, h2, H, B, H, H, fn2_b, nullptr, 0);
    head_update_kernel<<<(B * C + 255) / 256, 256, 0, stream>>>(
        h2, mu_W, mu_b, var_W, var_b,
        eps + (size_t)s * B * C, yt, (float*)d_out + (size_t)s * B * C,
        B, H, C);
  }
}
